// HybridGNNRNN_14413910245709
// MI455X (gfx1250) — matrix operand order repaired and checked by reading, not compiled, not run on hardware
//
#include <hip/hip_runtime.h>

// ---------------------------------------------------------------------------
// Problem constants (from reference)
// ---------------------------------------------------------------------------
#define N_NODES 102400
#define N_EDGES 1638400
#define IN_DIM  16
#define GNN_H   64
#define RNN_H   128
#define GATES   512      // 4 * RNN_H
#define OUT_DIM 7
#define SEQ_LEN 50
#define BATCH   2048     // N_NODES / SEQ_LEN

typedef __attribute__((ext_vector_type(16))) __bf16 bf16x16;  // 32 B
typedef __attribute__((ext_vector_type(8)))  __bf16 bf16x8;   // 16 B
typedef __attribute__((ext_vector_type(8)))  float  f32x8;

// ---------------------------------------------------------------------------
// WMMA helper (CDNA5 wave32, 16x16x32 bf16 -> f32)
// ---------------------------------------------------------------------------
__device__ __forceinline__ f32x8 wmma_bf16(bf16x16 a, bf16x16 b, f32x8 c) {
  return __builtin_amdgcn_wmma_f32_16x16x32_bf16(
      false, a, false, b, (short)0, c, false, false);
}

// A fragment: 16x32 (MxK) row-major, lda multiple of 8, rows 16B-aligned.
// lane L: m=L%16, kh=L/16; elems 0..7 = K kh*8+0..7 ; elems 8..15 = K 16+kh*8+0..7
// Two 16-byte vector loads per lane (KV==16 zero-pads the upper half).
template <int KV>
__device__ __forceinline__ bf16x16 load_A16(const __bf16* A, int lda, int lane) {
  const int m  = lane & 15;
  const int kh = (lane >> 4) & 1;
  const __bf16* row = A + (size_t)m * lda + kh * 8;
  bf16x16 a;
  const bf16x8 lo = *reinterpret_cast<const bf16x8*>(row);
#pragma unroll
  for (int i = 0; i < 8; ++i) a[i] = lo[i];
  if (KV > 16) {
    const bf16x8 hi = *reinterpret_cast<const bf16x8*>(row + 16);
#pragma unroll
    for (int i = 0; i < 8; ++i) a[i + 8] = hi[i];
  } else {
#pragma unroll
    for (int i = 0; i < 8; ++i) a[i + 8] = (__bf16)0.0f;
  }
  return a;
}

// B fragment from pre-packed fragment-major layout: [frag][lane][16] bf16.
// One aligned 32-byte vector load per lane (2x global_load_b128).
__device__ __forceinline__ bf16x16 load_B_packed(const __bf16* Bp, int frag, int lane) {
  return *reinterpret_cast<const bf16x16*>(Bp + ((size_t)frag * 32 + lane) * 16);
}

__device__ __forceinline__ float sigmoidf_(float x) { return 1.0f / (1.0f + __expf(-x)); }
__device__ __forceinline__ float tanhf_(float x) {
  const float e = __expf(-2.0f * x);
  return (1.0f - e) / (1.0f + e);
}

// ---------------------------------------------------------------------------
// Weight packing: build fragment-major bf16 B tiles, zero-padded.
// dst[((nt*KT + kt)*32 + lane)*16 + i],  k = kt*32 + (i<8 ? (lane>>4)*8 + i : 16 + (lane>>4)*8 + i-8),
// n = nt*16 + (lane&15).
// ---------------------------------------------------------------------------
__global__ void pack_B_NK_kernel(const float* __restrict__ W,  // [N][K] (torch: B = W^T)
                                 __bf16* __restrict__ dst, int K, int N, int KT, int NT) {
  const int idx = blockIdx.x * blockDim.x + threadIdx.x;
  if (idx >= NT * KT * 512) return;
  const int i    = idx & 15;
  const int lane = (idx >> 4) & 31;
  const int frag = idx >> 9;
  const int kt = frag % KT, nt = frag / KT;
  const int k = kt * 32 + ((lane >> 4) & 1) * 8 + (i & 7) + (i >> 3) * 16;
  const int n = nt * 16 + (lane & 15);
  dst[idx] = (k < K && n < N) ? (__bf16)W[(size_t)n * K + k] : (__bf16)0.0f;
}

__global__ void pack_B_KN_kernel(const float* __restrict__ W,  // [K][N]
                                 __bf16* __restrict__ dst, int K, int N, int KT, int NT) {
  const int idx = blockIdx.x * blockDim.x + threadIdx.x;
  if (idx >= NT * KT * 512) return;
  const int i    = idx & 15;
  const int lane = (idx >> 4) & 31;
  const int frag = idx >> 9;
  const int kt = frag % KT, nt = frag / KT;
  const int k = kt * 32 + ((lane >> 4) & 1) * 8 + (i & 7) + (i >> 3) * 16;
  const int n = nt * 16 + (lane & 15);
  dst[idx] = (k < K && n < N) ? (__bf16)W[(size_t)k * N + n] : (__bf16)0.0f;
}

// ---------------------------------------------------------------------------
// Small elementwise / prep kernels
// ---------------------------------------------------------------------------
__global__ void cvt_bf16_kernel(const float* __restrict__ s, __bf16* __restrict__ d, int n) {
  const int i = blockIdx.x * blockDim.x + threadIdx.x;
  if (i < n) d[i] = (__bf16)s[i];
}

__global__ void add_vec_kernel(const float* __restrict__ a, const float* __restrict__ b,
                               float* __restrict__ o, int n) {
  const int i = blockIdx.x * blockDim.x + threadIdx.x;
  if (i < n) o[i] = a[i] + b[i];
}

__global__ void deg_init_kernel(float* __restrict__ deg, int n) {
  const int i = blockIdx.x * blockDim.x + threadIdx.x;
  if (i < n) deg[i] = 1.0f;   // self-loop
}

__global__ void deg_accum_kernel(const int* __restrict__ dst, float* __restrict__ deg, int ne) {
  const int i = blockIdx.x * blockDim.x + threadIdx.x;
  if (i < ne) atomicAdd(&deg[dst[i]], 1.0f);
}

__global__ void deg_rsqrt_kernel(float* __restrict__ deg, int n) {
  const int i = blockIdx.x * blockDim.x + threadIdx.x;
  if (i < n) deg[i] = rsqrtf(deg[i]);   // deg >= 1 always
}

__global__ void self_loop_kernel(const float* __restrict__ h, const float* __restrict__ dinv,
                                 float* __restrict__ out, int total) {
  const int idx = blockIdx.x * blockDim.x + threadIdx.x;
  if (idx >= total) return;
  const int i = idx >> 6;               // GNN_H == 64
  const float w = dinv[i];
  out[idx] = h[idx] * w * w;
}

__global__ void edge_scatter_kernel(const int* __restrict__ src, const int* __restrict__ dst,
                                    const float* __restrict__ h, const float* __restrict__ dinv,
                                    float* __restrict__ out, int ne) {
  const int t = blockIdx.x * blockDim.x + threadIdx.x;
  const int e = t >> 4;
  if (e >= ne) return;
  const int j = (t & 15) * 4;
  const int s = src[e], d = dst[e];
  const float w = dinv[s] * dinv[d];
  const float4 hv = *reinterpret_cast<const float4*>(h + (size_t)s * GNN_H + j);
  float* o = out + (size_t)d * GNN_H + j;
  atomicAdd(o + 0, hv.x * w);
  atomicAdd(o + 1, hv.y * w);
  atomicAdd(o + 2, hv.z * w);
  atomicAdd(o + 3, hv.w * w);
}

__global__ void bias_relu_bf16_kernel(const float* __restrict__ in, const float* __restrict__ bias,
                                      __bf16* __restrict__ out, int total) {
  const int idx = blockIdx.x * blockDim.x + threadIdx.x;
  if (idx >= total) return;
  const float v = in[idx] + bias[idx & (GNN_H - 1)];
  out[idx] = (__bf16)fmaxf(v, 0.0f);
}

// ---------------------------------------------------------------------------
// GCN dense GEMM: out[M][N] = A[M][K] @ B[K][N]; A bf16 row-major, B packed.
// 4 waves/block, one 16-row M-tile per wave.
// ---------------------------------------------------------------------------
template <int K, int N>
__global__ void gcn_gemm_kernel(const __bf16* __restrict__ A, const __bf16* __restrict__ Bp,
                                float* __restrict__ out, int M) {
  constexpr int KT = (K + 31) / 32;
  const int wave = threadIdx.x >> 5;
  const int lane = threadIdx.x & 31;
  const int row0 = (blockIdx.x * 4 + wave) * 16;
  if (row0 >= M) return;                       // wave-uniform
  const int n = lane & 15, half = lane >> 4;
  bf16x16 a[KT];
#pragma unroll
  for (int kt = 0; kt < KT; ++kt) {
    const __bf16* base = A + (size_t)row0 * K + kt * 32;
    a[kt] = (K - kt * 32 >= 32) ? load_A16<32>(base, K, lane)
                                : load_A16<16>(base, K, lane);
  }
#pragma unroll
  for (int nt = 0; nt < N / 16; ++nt) {
    f32x8 acc = {};
#pragma unroll
    for (int kt = 0; kt < KT; ++kt)
      acc = wmma_bf16(a[kt], load_B_packed(Bp, nt * KT + kt, lane), acc);
#pragma unroll
    for (int v = 0; v < 8; ++v)
      out[(size_t)(row0 + v + 8 * half) * N + nt * 16 + n] = acc[v];
  }
}

// ---------------------------------------------------------------------------
// Fused 2-layer LSTM + FC. One block (4 waves) owns one 16-row batch tile.
// Each wave computes hidden-column tiles jt = {wave, wave+4}; h ping-pongs in
// LDS (statically indexed -> ds ops); c lives in VGPRs; weights are packed
// fragment-major in L2.
// ---------------------------------------------------------------------------
template <int XKT>  // k-tiles of the x-input (2 for layer0, 4 for layer1)
__device__ __forceinline__ void lstm_step(const __bf16* xsrc, int ldx,
                                          const __bf16* __restrict__ wxp,
                                          const __bf16* __restrict__ whp,
                                          const float* __restrict__ bias,
                                          const __bf16* hprev, __bf16* hnext,
                                          f32x8* c, int lane, int wave) {
  const int n = lane & 15, half = lane >> 4;
  bf16x16 ax[XKT];
#pragma unroll
  for (int kt = 0; kt < XKT; ++kt)
    ax[kt] = load_A16<32>(xsrc + kt * 32, ldx, lane);
  bf16x16 ah[4];
#pragma unroll
  for (int kt = 0; kt < 4; ++kt)
    ah[kt] = load_A16<32>(hprev + kt * 32, RNN_H, lane);

#pragma unroll
  for (int jj = 0; jj < 2; ++jj) {
    const int jt = wave + jj * 4;          // this wave's hidden-column tile
    f32x8 g[4];
#pragma unroll
    for (int gi = 0; gi < 4; ++gi) {       // torch gate order i, f, g, o
      const int nt = gi * 8 + jt;          // tile index into the 512-wide gates
      const float bv = bias[nt * 16 + n];
      f32x8 acc;
#pragma unroll
      for (int v = 0; v < 8; ++v) acc[v] = bv;
#pragma unroll
      for (int kt = 0; kt < XKT; ++kt)
        acc = wmma_bf16(ax[kt], load_B_packed(wxp, nt * XKT + kt, lane), acc);
#pragma unroll
      for (int kt = 0; kt < 4; ++kt)
        acc = wmma_bf16(ah[kt], load_B_packed(whp, nt * 4 + kt, lane), acc);
      g[gi] = acc;
    }
#pragma unroll
    for (int v = 0; v < 8; ++v) {
      const float iv = sigmoidf_(g[0][v]);
      const float fv = sigmoidf_(g[1][v]);
      const float gv = tanhf_(g[2][v]);
      const float ov = sigmoidf_(g[3][v]);
      const float cv = fv * c[jj][v] + iv * gv;
      c[jj][v] = cv;
      hnext[(v + 8 * half) * RNN_H + jt * 16 + n] = (__bf16)(ov * tanhf_(cv));
    }
  }
}

__global__ void lstm_fc_kernel(const __bf16* __restrict__ xin,    // [B*T][64] node-major
                               const __bf16* __restrict__ w0xp,   // packed [32nt][2kt]
                               const __bf16* __restrict__ w0hp,   // packed [32nt][4kt]
                               const float* __restrict__ bias0,   // [512]
                               const __bf16* __restrict__ w1xp,   // packed [32nt][4kt]
                               const __bf16* __restrict__ w1hp,   // packed [32nt][4kt]
                               const float* __restrict__ bias1,   // [512]
                               const __bf16* __restrict__ fcwp,   // packed [1nt][4kt]
                               const float* __restrict__ fcb,     // [7]
                               float* __restrict__ out) {         // [B][7]
  __shared__ __attribute__((aligned(16))) __bf16 h0s[2][16 * RNN_H];
  __shared__ __attribute__((aligned(16))) __bf16 h1s[2][16 * RNN_H];
  const int wave = threadIdx.x >> 5;
  const int lane = threadIdx.x & 31;
  const int b0 = blockIdx.x * 16;

  for (int i = threadIdx.x; i < 16 * RNN_H; i += 128) {
    h0s[0][i] = (__bf16)0.0f;
    h1s[0][i] = (__bf16)0.0f;
  }
  __syncthreads();

  f32x8 c0[2], c1[2];
#pragma unroll
  for (int j = 0; j < 2; ++j)
#pragma unroll
    for (int v = 0; v < 8; ++v) { c0[j][v] = 0.0f; c1[j][v] = 0.0f; }

  // Anti-hoist trick, v2: launder a byte *offset* (always 0) through an opaque
  // asm instead of the pointers themselves.  The weight loads then depend on a
  // per-iteration unknown value (no hoist across timesteps -> no spills), but
  // the base pointers remain provably GLOBAL, so loads lower to
  // global_load_b128 (saddr+voffset) instead of flat_load_b128 — no DScnt
  // coupling, no 64-bit VGPR address pairs (round-2 failure mode).
  size_t ah_off = 0;

  for (int tt = 0; tt < SEQ_LEN / 2; ++tt) {
    asm volatile("" : "+s"(ah_off));
    const __bf16* w0x = w0xp + ah_off;
    const __bf16* w0h = w0hp + ah_off;
    const __bf16* w1x = w1xp + ah_off;
    const __bf16* w1h = w1hp + ah_off;

    {  // t = 2*tt : buffers 0 -> 1
      const __bf16* xt = xin + ((size_t)b0 * SEQ_LEN + 2 * tt) * GNN_H;
      lstm_step<2>(xt, SEQ_LEN * GNN_H, w0x, w0h, bias0, h0s[0], h0s[1], c0, lane, wave);
      __syncthreads();
      lstm_step<4>(h0s[1], RNN_H, w1x, w1h, bias1, h1s[0], h1s[1], c1, lane, wave);
      __syncthreads();
    }
    {  // t = 2*tt + 1 : buffers 1 -> 0
      const __bf16* xt = xin + ((size_t)b0 * SEQ_LEN + 2 * tt + 1) * GNN_H;
      lstm_step<2>(xt, SEQ_LEN * GNN_H, w0x, w0h, bias0, h0s[1], h0s[0], c0, lane, wave);
      __syncthreads();
      lstm_step<4>(h0s[0], RNN_H, w1x, w1h, bias1, h1s[1], h1s[0], c1, lane, wave);
      __syncthreads();
    }
  }

  // FC head on final h1 (in buffer 0 after 50 steps). Wave 0 only (uniform).
  if (wave == 0) {
    const int n = lane & 15, half = lane >> 4;
    f32x8 acc;
    const float bv = (n < OUT_DIM) ? fcb[n] : 0.0f;
#pragma unroll
    for (int v = 0; v < 8; ++v) acc[v] = bv;
#pragma unroll
    for (int kt = 0; kt < 4; ++kt) {
      bf16x16 a = load_A16<32>(&h1s[0][kt * 32], RNN_H, lane);
      acc = wmma_bf16(a, load_B_packed(fcwp, kt, lane), acc);
    }
    if (n < OUT_DIM) {
#pragma unroll
      for (int v = 0; v < 8; ++v)
        out[(size_t)(b0 + v + 8 * half) * OUT_DIM + n] = acc[v];
    }
  }
}

// ---------------------------------------------------------------------------
// Host-side orchestration
// ---------------------------------------------------------------------------
extern "C" void kernel_launch(void* const* d_in, const int* in_sizes, int n_in,
                              void* d_out, int out_size, void* d_ws, size_t ws_size,
                              hipStream_t stream) {
  (void)in_sizes; (void)n_in; (void)out_size; (void)ws_size;

  const float* x    = (const float*)d_in[0];
  const int*   ei   = (const int*)d_in[1];     // [2][N_EDGES]: row0=src, row1=dst
  const float* w1   = (const float*)d_in[3];
  const float* b1   = (const float*)d_in[4];
  const float* w2   = (const float*)d_in[5];
  const float* b2   = (const float*)d_in[6];
  const float* wih0 = (const float*)d_in[7];
  const float* whh0 = (const float*)d_in[8];
  const float* bih0 = (const float*)d_in[9];
  const float* bhh0 = (const float*)d_in[10];
  const float* wih1 = (const float*)d_in[11];
  const float* whh1 = (const float*)d_in[12];
  const float* bih1 = (const float*)d_in[13];
  const float* bhh1 = (const float*)d_in[14];
  const float* fcw  = (const float*)d_in[15];
  const float* fcb  = (const float*)d_in[16];
  float* out = (float*)d_out;

  const int* srcp = ei;
  const int* dstp = ei + N_EDGES;

  // ---- workspace carve-out (256B aligned) ----
  char* ws = (char*)d_ws;
  size_t cur = 0;
  auto wsalloc = [&](size_t bytes) -> void* {
    void* p = ws + cur;
    cur = (cur + bytes + 255) & ~(size_t)255;
    return p;
  };
  __bf16* xb    = (__bf16*)wsalloc((size_t)N_NODES * IN_DIM * 2);
  __bf16* w1p   = (__bf16*)wsalloc((size_t)4 * 1 * 512 * 2);    // [4nt][1kt]
  __bf16* w2p   = (__bf16*)wsalloc((size_t)4 * 2 * 512 * 2);    // [4nt][2kt]
  __bf16* w0xp  = (__bf16*)wsalloc((size_t)32 * 2 * 512 * 2);   // [32nt][2kt]
  __bf16* w0hp  = (__bf16*)wsalloc((size_t)32 * 4 * 512 * 2);   // [32nt][4kt]
  __bf16* w1xp  = (__bf16*)wsalloc((size_t)32 * 4 * 512 * 2);
  __bf16* w1hp  = (__bf16*)wsalloc((size_t)32 * 4 * 512 * 2);
  __bf16* fcwp  = (__bf16*)wsalloc((size_t)1 * 4 * 512 * 2);    // [1nt][4kt]
  float*  bias0 = (float*)wsalloc(GATES * 4);
  float*  bias1 = (float*)wsalloc(GATES * 4);
  float*  dinv  = (float*)wsalloc((size_t)N_NODES * 4);
  float*  hbuf  = (float*)wsalloc((size_t)N_NODES * GNN_H * 4);   // GEMM out
  float*  agg   = (float*)wsalloc((size_t)N_NODES * GNN_H * 4);   // scatter out
  __bf16* hb    = (__bf16*)wsalloc((size_t)N_NODES * GNN_H * 2);  // relu'd bf16

  auto cdiv = [](long a, long b) { return (int)((a + b - 1) / b); };
  const int TPB = 256;
  const int TOTAL = N_NODES * GNN_H;

  // ---- weight prep: bf16 convert + fragment-major packing ----
  cvt_bf16_kernel<<<cdiv(N_NODES * IN_DIM, TPB), TPB, 0, stream>>>(x, xb, N_NODES * IN_DIM);
  pack_B_KN_kernel<<<cdiv(4L * 1 * 512, TPB), TPB, 0, stream>>>(w1, w1p, IN_DIM, GNN_H, 1, 4);
  pack_B_KN_kernel<<<cdiv(4L * 2 * 512, TPB), TPB, 0, stream>>>(w2, w2p, GNN_H, GNN_H, 2, 4);
  pack_B_NK_kernel<<<cdiv(32L * 2 * 512, TPB), TPB, 0, stream>>>(wih0, w0xp, GNN_H, GATES, 2, 32);
  pack_B_NK_kernel<<<cdiv(32L * 4 * 512, TPB), TPB, 0, stream>>>(whh0, w0hp, RNN_H, GATES, 4, 32);
  pack_B_NK_kernel<<<cdiv(32L * 4 * 512, TPB), TPB, 0, stream>>>(wih1, w1xp, RNN_H, GATES, 4, 32);
  pack_B_NK_kernel<<<cdiv(32L * 4 * 512, TPB), TPB, 0, stream>>>(whh1, w1hp, RNN_H, GATES, 4, 32);
  pack_B_KN_kernel<<<cdiv(1L * 4 * 512, TPB), TPB, 0, stream>>>(fcw, fcwp, RNN_H, OUT_DIM, 4, 1);
  add_vec_kernel<<<cdiv(GATES, TPB), TPB, 0, stream>>>(bih0, bhh0, bias0, GATES);
  add_vec_kernel<<<cdiv(GATES, TPB), TPB, 0, stream>>>(bih1, bhh1, bias1, GATES);

  // ---- symmetric normalization factors ----
  deg_init_kernel<<<cdiv(N_NODES, TPB), TPB, 0, stream>>>(dinv, N_NODES);
  deg_accum_kernel<<<cdiv(N_EDGES, TPB), TPB, 0, stream>>>(dstp, dinv, N_EDGES);
  deg_rsqrt_kernel<<<cdiv(N_NODES, TPB), TPB, 0, stream>>>(dinv, N_NODES);

  // ---- GCN layer 1 ----
  gcn_gemm_kernel<IN_DIM, GNN_H><<<N_NODES / 64, 128, 0, stream>>>(xb, w1p, hbuf, N_NODES);
  self_loop_kernel<<<cdiv(TOTAL, TPB), TPB, 0, stream>>>(hbuf, dinv, agg, TOTAL);
  edge_scatter_kernel<<<cdiv((long)N_EDGES * 16, TPB), TPB, 0, stream>>>(srcp, dstp, hbuf, dinv, agg, N_EDGES);
  bias_relu_bf16_kernel<<<cdiv(TOTAL, TPB), TPB, 0, stream>>>(agg, b1, hb, TOTAL);

  // ---- GCN layer 2 ----
  gcn_gemm_kernel<GNN_H, GNN_H><<<N_NODES / 64, 128, 0, stream>>>(hb, w2p, hbuf, N_NODES);
  self_loop_kernel<<<cdiv(TOTAL, TPB), TPB, 0, stream>>>(hbuf, dinv, agg, TOTAL);
  edge_scatter_kernel<<<cdiv((long)N_EDGES * 16, TPB), TPB, 0, stream>>>(srcp, dstp, hbuf, dinv, agg, N_EDGES);
  bias_relu_bf16_kernel<<<cdiv(TOTAL, TPB), TPB, 0, stream>>>(agg, b2, hb, TOTAL);

  // ---- fused 2-layer LSTM (50 steps) + FC head, single launch ----
  lstm_fc_kernel<<<BATCH / 16, 128, 0, stream>>>(
      hb, w0xp, w0hp, bias0, w1xp, w1hp, bias1, fcwp, fcb, out);
}
